// EventDrivenSparseFoveatedSNN_27616639713860
// MI455X (gfx1250) — matrix operand order repaired and checked by reading, not compiled, not run on hardware
//
#include <hip/hip_runtime.h>
#include <hip/hip_bf16.h>
#include <math.h>

// ---------------------------------------------------------------------------
// EventDrivenSparseFoveatedSNN for MI455X (gfx1250, wave32, WMMA).
// Convs + head are implicit-im2col / packed GEMMs using v_wmma_f32_16x16x32_f16
// (f16 in, f32 accum). Problem is HBM-bound (~134MB x_seq @ 23.3TB/s ~ 6us),
// so we stage compact f16 windows per timestep (async-to-LDS where available)
// and keep everything else in L2/LDS.
// ---------------------------------------------------------------------------

typedef __attribute__((ext_vector_type(16))) _Float16 v16h;
typedef __attribute__((ext_vector_type(8)))  float    v8f;
typedef __attribute__((ext_vector_type(4)))  int      v4i;

#define TT 32
#define BB 32
#define HH 128
#define WW 128
#define CC 2
#define FHW 32          // fovea window 32x32
#define CF 64
#define CP 32
#define NOUT 128
#define NPATCH 256      // 16x16 patches
#define HP 64           // periph 64x64
#define KCAT 352        // 96 (fused) + 256 (scores), = 11*32
#define EVT 0.5f
#define BETA 0.9f

// gfx1250 async global->LDS path (ASYNCcnt). Guarded: falls back to plain
// vectorized LDS stores if the toolchain lacks the builtins.
#if defined(__HIP_DEVICE_COMPILE__) && __has_builtin(__builtin_amdgcn_global_load_async_to_lds_b128)
#define USE_ASYNC_LDS 1
#else
#define USE_ASYNC_LDS 0
#endif

__device__ __forceinline__ void wait_async_lds() {
#if USE_ASYNC_LDS
#if __has_builtin(__builtin_amdgcn_s_wait_asynccnt)
  __builtin_amdgcn_s_wait_asynccnt(0);
#else
  asm volatile("s_wait_asynccnt 0x0" ::: "memory");
#endif
#endif
}

// ---------------- init: zero state/accums, convert+pack weights to f16 ------
__global__ void init_ws_k(float* vf, float* vp, float* accums,
                          const float* w_f, const float* w_p,
                          const float* head_w, const float* route_w,
                          _Float16* wf16, _Float16* wp16, _Float16* wcat16) {
  size_t idx = (size_t)blockIdx.x * 256 + threadIdx.x;
  if (idx < (size_t)BB * 1024 * CF) vf[idx] = 0.f;
  if (idx < (size_t)BB * 4096 * CP) vp[idx] = 0.f;
  if (idx < 512) accums[idx] = 0.f;
  if (idx < 32 * CF) {               // padded (K=32, N=64): flat k*64+n matches HWIO flat
    int k = (int)idx >> 6;
    wf16[idx] = (_Float16)((k < 18) ? w_f[idx] : 0.f);
  }
  if (idx < 32 * CP) {
    int k = (int)idx >> 5;
    wp16[idx] = (_Float16)((k < 18) ? w_p[idx] : 0.f);
  }
  if (idx < (size_t)KCAT * NOUT) {   // concat [head_w; route_w] as (352,128) f16
    int k = (int)idx >> 7, n = (int)idx & 127;
    wcat16[idx] = (_Float16)((k < CF + CP) ? head_w[idx]
                                           : route_w[(size_t)(k - (CF + CP)) * NOUT + n]);
  }
}

// ---------------- per-t: patch scores, top-4 gate, argmax, active ratio -----
__global__ void prep_scores_k(const float* __restrict__ x_t, float* scores, float* fused,
                              int* y0a, int* x0a, float* route_accum, float* active_accum) {
  int b = blockIdx.x;
  int p = threadIdx.x;                 // patch id 0..255
  int py = p >> 4, px = p & 15;
  const float* xb = x_t + (size_t)b * HH * WW * CC;
  float s = 0.f; int cnt = 0;
  for (int dy = 0; dy < 8; ++dy)
    for (int dx = 0; dx < 8; ++dx)
      for (int c = 0; c < 2; ++c) {
        float v = xb[((py * 8 + dy) * WW + (px * 8 + dx)) * CC + c];
        float av = fabsf(v);
        if (av > EVT) { s += av; cnt++; }
      }
  s *= (1.f / 128.f);
  scores[b * NPATCH + p] = s;
  if (p < CF + CP) fused[b * (CF + CP) + p] = 0.f;   // zero fused accumulators for this t

  __shared__ int   scnt[256];
  __shared__ float work[256];
  __shared__ float rv[256];
  __shared__ int   ri[256];
  scnt[p] = cnt; work[p] = s;
  __syncthreads();
  for (int off = 128; off > 0; off >>= 1) {
    if (p < off) scnt[p] += scnt[p + off];
    __syncthreads();
  }
  if (p == 0) atomicAdd(active_accum, (float)scnt[0]);

  float thresh = 0.f; int best = 0;
  for (int round = 0; round < 4; ++round) {
    rv[p] = work[p]; ri[p] = p;
    __syncthreads();
    for (int off = 128; off > 0; off >>= 1) {
      if (p < off) {
        if (rv[p + off] > rv[p] || (rv[p + off] == rv[p] && ri[p + off] < ri[p])) {
          rv[p] = rv[p + off]; ri[p] = ri[p + off];
        }
      }
      __syncthreads();
    }
    thresh = rv[0];
    int widx = ri[0];
    if (round == 0 && p == 0) best = widx;
    __syncthreads();
    if (p == widx) work[p] = -1e30f;
    __syncthreads();
  }
  float gate = (s >= thresh) ? 1.f : 0.f;
  atomicAdd(&route_accum[p], gate);
  if (p == 0) {
    int cy = (best >> 4) * 8 + 4, cx = (best & 15) * 8 + 4;
    int y0 = min(max(cy - 16, 0), HH - FHW);
    int x0 = min(max(cx - 16, 0), WW - FHW);
    y0a[b] = y0; x0a[b] = x0;
  }
}

// ---------------- per-t: f16 staging of fovea window and 1/2 periphery ------
__global__ void prep_stage_k(const float* __restrict__ x_t, const int* y0a, const int* x0a,
                             _Float16* fov16, _Float16* per16) {
  int idx = blockIdx.x * 256 + threadIdx.x;
  const int NF = BB * 2048;
  if (idx < NF) {
    int b = idx >> 11, r = idx & 2047;
    int y = r >> 6, x = (r >> 1) & 31, c = r & 1;
    int gy = y0a[b] + y, gx = x0a[b] + x;
    float v = x_t[((size_t)b * HH * WW + gy * WW + gx) * CC + c];
    if (!(fabsf(v) > EVT)) v = 0.f;
    fov16[idx] = (_Float16)v;
  } else {
    int j = idx - NF;
    if (j < BB * 8192) {
      int b = j >> 13, r = j & 8191;
      int i = r >> 7, jx = (r >> 1) & 63, c = r & 1;
      const float* xb = x_t + (size_t)b * HH * WW * CC;
      float acc = 0.f;
      for (int dy = 0; dy < 2; ++dy)
        for (int dx = 0; dx < 2; ++dx) {
          float v = xb[((2 * i + dy) * WW + (2 * jx + dx)) * CC + c];
          if (fabsf(v) > EVT) acc += v;
        }
      per16[j] = (_Float16)(0.25f * acc);  // half-pixel-center bilinear == 2x2 mean
    }
  }
}

// ---------------- fovea conv: GEMM (B*1024 x 18pad32) x (32 x 64) -----------
// Block = 128 threads = 4 waves; one 16-row M tile per block, wave w owns N
// tile [16w,16w+16). A built in contiguous k order (half h holds k 16h..16h+15), B per lane-striped
// KxN layout, C/D per documented 8-VGPR layout.
__global__ void conv_fovea_k(const _Float16* __restrict__ fov16,
                             const _Float16* __restrict__ wf16,
                             float* __restrict__ ff) {
  int blk = blockIdx.x;
  int b = blk >> 6;            // 64 m-tiles per batch (1024 positions)
  int mt = blk & 63;
  int wave = threadIdx.x >> 5; // N tile 0..3
  int lane = threadIdx.x & 31;

  __shared__ __align__(16) _Float16 win[2048];   // full 32x32x2 window (4KB)
  const _Float16* fb = fov16 + (size_t)b * 2048;
#if USE_ASYNC_LDS
  // 4KB window = 256 x b128 lane-transfers via the ASYNCcnt-tracked DMA path.
  for (int i = threadIdx.x; i < 256; i += 128) {
    __builtin_amdgcn_global_load_async_to_lds_b128(
        (__attribute__((address_space(1))) v4i*)((const char*)fb + i * 16),
        (__attribute__((address_space(3))) v4i*)((char*)win + i * 16), 0, 0);
  }
  wait_async_lds();
#else
  for (int i = threadIdx.x; i < 256; i += 128)
    ((int4*)win)[i] = ((const int4*)fb)[i];
#endif
  __syncthreads();

  int ml = lane & 15;
  int m = mt * 16 + ml;                // output position (same row for tile)
  int oy = m >> 5, ox = m & 31;
  int kb0 = (lane < 16) ? 0 : 16;      // A: lanes 0-15 K 0-15, lanes 16-31 K 16-31
  v16h a;
#pragma unroll
  for (int i = 0; i < 16; ++i) {
    int k = kb0 + i;
    _Float16 v = (_Float16)0;
    if (k < 18) {
      int ky = k / 6, rem = k % 6;
      int kx = rem >> 1, c = rem & 1;
      int gy = oy + ky - 1, gx = ox + kx - 1;
      if (gy >= 0 && gy < 32 && gx >= 0 && gx < 32)
        v = win[(gy << 6) + (gx << 1) + c];
    }
    a[i] = v;
  }

  int n = wave * 16 + ml;
  int kb = (lane < 16) ? 0 : 16;       // B: lanes 0-15 K 0-15, lanes 16-31 K 16-31
  v16h bv;
#pragma unroll
  for (int i = 0; i < 16; ++i) bv[i] = wf16[(kb + i) * CF + n];

  v8f c8 = {};
  c8 = __builtin_amdgcn_wmma_f32_16x16x32_f16(false, a, false, bv,
                                              (short)0, c8, false, false);

  int m0 = mt * 16 + ((lane < 16) ? 0 : 8);
  float* outp = ff + ((size_t)b * 1024 + m0) * CF + n;   // rows are +256B apart
#pragma unroll
  for (int r = 0; r < 8; ++r)
    outp[r * CF] = c8[r];
}

// ---------------- periph conv: GEMM (B*4096 x 18pad32) x (32 x 32) ----------
// Block = 64 threads = 2 waves; stage only the 3 input rows this tile needs
// (zero-filled at image edges, so plain LDS staging here).
__global__ void conv_periph_k(const _Float16* __restrict__ per16,
                              const _Float16* __restrict__ wp16,
                              float* __restrict__ pf) {
  int blk = blockIdx.x;
  int b = blk >> 8;            // 256 m-tiles per batch (4096 positions)
  int mt = blk & 255;
  int wave = threadIdx.x >> 5; // N tile 0..1
  int lane = threadIdx.x & 31;
  int oy = mt >> 2;            // output row (4 tiles per 64-wide row)
  int oxb = (mt & 3) * 16;

  __shared__ _Float16 win[3 * 128];    // rows oy-1..oy+1, 64 cols x 2 ch
  const _Float16* pb = per16 + (size_t)b * 8192;
  for (int i = threadIdx.x; i < 384; i += 64) {
    int rr = i / 128, cc = i % 128;
    int gy = oy + rr - 1;
    win[i] = (gy >= 0 && gy < HP) ? pb[gy * 128 + cc] : (_Float16)0;
  }
  __syncthreads();

  int ml = lane & 15;
  int kb0 = (lane < 16) ? 0 : 16;
  v16h a;
#pragma unroll
  for (int i = 0; i < 16; ++i) {
    int k = kb0 + i;
    _Float16 v = (_Float16)0;
    if (k < 18) {
      int ky = k / 6, rem = k % 6;
      int kx = rem >> 1, c = rem & 1;
      int gx = oxb + ml + kx - 1;
      if (gx >= 0 && gx < HP) v = win[ky * 128 + (gx << 1) + c];
    }
    a[i] = v;
  }

  int n = wave * 16 + ml;
  int kb = (lane < 16) ? 0 : 16;
  v16h bv;
#pragma unroll
  for (int i = 0; i < 16; ++i) bv[i] = wp16[(kb + i) * CP + n];

  v8f c8 = {};
  c8 = __builtin_amdgcn_wmma_f32_16x16x32_f16(false, a, false, bv,
                                              (short)0, c8, false, false);

  int m0 = mt * 16 + ((lane < 16) ? 0 : 8);
  float* outp = pf + ((size_t)b * 4096 + m0) * CP + n;   // rows are +128B apart
#pragma unroll
  for (int r = 0; r < 8; ++r)
    outp[r * CP] = c8[r];
}

// ---------------- per-t: channel saliency + kWTA-16 gate --------------------
__global__ void sal_gate_k(const float* __restrict__ ff, float* gate, float* chan_accum) {
  int b = blockIdx.x;
  int ch = threadIdx.x;                // 64 channels
  const float* fb = ff + (size_t)b * 1024 * CF;
  float s = 0.f;
  for (int p = 0; p < 1024; ++p) {
    __builtin_prefetch(&fb[(p + 8) * CF + ch], 0, 1);
    s += fabsf(fb[p * CF + ch]);
  }
  s *= (1.f / 1024.f);
  __shared__ float work[64], rv[64];
  __shared__ int ri[64];
  work[ch] = s;
  __syncthreads();
  float thresh = 0.f;
  for (int round = 0; round < 16; ++round) {
    rv[ch] = work[ch]; ri[ch] = ch;
    __syncthreads();
    for (int off = 32; off > 0; off >>= 1) {
      if (ch < off) {
        if (rv[ch + off] > rv[ch] || (rv[ch + off] == rv[ch] && ri[ch + off] < ri[ch])) {
          rv[ch] = rv[ch + off]; ri[ch] = ri[ch + off];
        }
      }
      __syncthreads();
    }
    thresh = rv[0];
    int widx = ri[0];
    __syncthreads();
    if (ch == widx) work[ch] = -1e30f;
    __syncthreads();
  }
  float g = (s >= thresh) ? 1.f : 0.f;
  gate[b * CF + ch] = g;
  atomicAdd(&chan_accum[ch], g);
}

// ---------------- per-t: LIF update + spike means ---------------------------
__global__ void lif_fovea_k(const float* __restrict__ ff, const float* __restrict__ gate,
                            float* vf, float* fused, float* sr_accum) {
  int idx = blockIdx.x * 256 + threadIdx.x;    // BB*1024*64
  int b = idx >> 16;
  int ch = idx & 63;
  float f = ff[idx] * gate[b * CF + ch];
  float v = BETA * vf[idx] + f;
  float s = (v - 1.0f > 0.f) ? 1.f : 0.f;
  vf[idx] = v - s;
  __shared__ float acc[64];
  int t = threadIdx.x;
  if (t < 64) acc[t] = 0.f;
  __syncthreads();
  if (s != 0.f) atomicAdd(&acc[ch], s);
  __syncthreads();
  if (t < 64 && acc[t] != 0.f) atomicAdd(&fused[b * (CF + CP) + t], acc[t] * (1.f / 1024.f));
  if (t == 0) {
    float bs = 0.f;
    for (int i = 0; i < 64; ++i) bs += acc[i];
    if (bs != 0.f) atomicAdd(sr_accum, bs);
  }
}

__global__ void lif_periph_k(const float* __restrict__ pf, float* vp,
                             float* fused, float* sr_accum) {
  int idx = blockIdx.x * 256 + threadIdx.x;    // BB*4096*32
  int b = idx >> 17;
  int ch = idx & 31;
  float v = BETA * vp[idx] + pf[idx];
  float s = (v - 1.0f > 0.f) ? 1.f : 0.f;
  vp[idx] = v - s;
  __shared__ float acc[32];
  int t = threadIdx.x;
  if (t < 32) acc[t] = 0.f;
  __syncthreads();
  if (s != 0.f) atomicAdd(&acc[ch], s);
  __syncthreads();
  if (t < 32 && acc[t] != 0.f) atomicAdd(&fused[b * (CF + CP) + CF + t], acc[t] * (1.f / 4096.f));
  if (t == 0) {
    float bs = 0.f;
    for (int i = 0; i < 32; ++i) bs += acc[i];
    if (bs != 0.f) atomicAdd(sr_accum, bs);
  }
}

// ---------------- per-t: head GEMM via WMMA ---------------------------------
// [fused(B x 96) | scores(B x 256)] @ wcat(352 x 128): M=32, K=352=11*32,
// N=128. One wave per 16x16 D tile (16 waves); K-loop of 11 WMMAs, A built by
// on-the-fly f32->f16 conversion of fused/scores.
__global__ void head_wmma_k(const float* __restrict__ fused, const float* __restrict__ scores,
                            const _Float16* __restrict__ wcat16,
                            const float* __restrict__ head_b, const float* __restrict__ route_b,
                            float* out, float* logits_seq, int t) {
  int tile = blockIdx.x;         // 0..15
  int mt = tile >> 3;            // 0..1
  int nt = tile & 7;             // 0..7
  int lane = threadIdx.x & 31;
  int ml = lane & 15;
  int brow = mt * 16 + ml;       // A row = batch index
  int kb0 = (lane < 16) ? 0 : 16;
  int n = nt * 16 + ml;
  int kbB = (lane < 16) ? 0 : 16;
  const float* fb = fused + brow * (CF + CP);
  const float* sb = scores + brow * NPATCH;

  v8f c8 = {};
#pragma unroll
  for (int kk = 0; kk < 11; ++kk) {
    int kbase = kk * 32;
    v16h a;
#pragma unroll
    for (int i = 0; i < 16; ++i) {
      int k = kbase + kb0 + i;
      float v = (k < CF + CP) ? fb[k] : sb[k - (CF + CP)];
      a[i] = (_Float16)v;
    }
    v16h bv;
#pragma unroll
    for (int i = 0; i < 16; ++i) bv[i] = wcat16[(size_t)(kbase + kbB + i) * NOUT + n];
    c8 = __builtin_amdgcn_wmma_f32_16x16x32_f16(false, a, false, bv,
                                                (short)0, c8, false, false);
  }

  float bias = head_b[n] + route_b[n];
  int b0 = mt * 16 + ((lane < 16) ? 0 : 8);
  float* lp = logits_seq + ((size_t)t * BB + b0) * NOUT + n;
  float* op = out + (size_t)b0 * NOUT + n;
#pragma unroll
  for (int r = 0; r < 8; ++r) {
    float l = c8[r] + bias;
    lp[r * NOUT] = l;
    float contrib = l * (1.f / (float)TT);
    if (t == 0) op[r * NOUT] = contrib;
    else        op[r * NOUT] += contrib;
  }
}

// ---------------- final statistics ------------------------------------------
__global__ void finalize_k(const float* accums, float* out_tail) {
  int i = threadIdx.x;
  if (i == 0) out_tail[0] = accums[0] / ((float)TT * BB * 1024.f * CF);
  if (i == 1) out_tail[1] = accums[1] / ((float)TT * BB * 4096.f * CP);
  if (i == 2) out_tail[2] = accums[2] / ((float)TT * BB * HH * WW * CC);
  if (i >= 3 && i < 3 + NPATCH)       out_tail[i] = accums[3 + (i - 3)] / ((float)TT * BB);
  if (i >= 3 + NPATCH && i < 3 + NPATCH + CF)
    out_tail[i] = accums[3 + NPATCH + (i - 3 - NPATCH)] / ((float)TT * BB);
}

// ---------------------------------------------------------------------------
extern "C" void kernel_launch(void* const* d_in, const int* in_sizes, int n_in,
                              void* d_out, int out_size, void* d_ws, size_t ws_size,
                              hipStream_t stream) {
  (void)in_sizes; (void)n_in; (void)out_size; (void)ws_size;
  const float* x_seq   = (const float*)d_in[0];
  const float* w_fovea = (const float*)d_in[1];
  const float* w_periph= (const float*)d_in[2];
  const float* head_w  = (const float*)d_in[3];
  const float* head_b  = (const float*)d_in[4];
  const float* route_w = (const float*)d_in[5];
  const float* route_b = (const float*)d_in[6];

  char* ws = (char*)d_ws;
  float*    vf      = (float*)(ws + 0);              // 8 MB
  float*    vp      = (float*)(ws + 8388608);        // 16 MB
  float*    ff      = (float*)(ws + 25165824);       // 8 MB
  float*    pf      = (float*)(ws + 33554432);       // 16 MB
  float*    scores  = (float*)(ws + 50331648);       // 32 KB
  float*    gate    = (float*)(ws + 50364416);       // 8 KB
  float*    fused   = (float*)(ws + 50372608);       // 12 KB
  int*      y0a     = (int*)  (ws + 50384896);
  int*      x0a     = (int*)  (ws + 50385024);
  float*    accums  = (float*)(ws + 50385152);       // [0]srF [1]srP [2]act [3..258]route [259..322]chan
  _Float16* wf16    = (_Float16*)(ws + 50387200);
  _Float16* wp16    = (_Float16*)(ws + 50391296);
  _Float16* fov16   = (_Float16*)(ws + 50393344);    // 128 KB
  _Float16* per16   = (_Float16*)(ws + 50524416);    // 512 KB
  _Float16* wcat16  = (_Float16*)(ws + 51048704);    // 88 KB (352x128 f16)

  float* out        = (float*)d_out;                 // (B,128)
  float* logits_seq = (float*)d_out + 4096;          // (T,B,128)
  float* out_tail   = (float*)d_out + 4096 + TT * BB * NOUT;  // sr(2), active(1), route(256), chan(64)

  init_ws_k<<<16384, 256, 0, stream>>>(vf, vp, accums, w_fovea, w_periph,
                                       head_w, route_w, wf16, wp16, wcat16);

  for (int t = 0; t < TT; ++t) {
    const float* x_t = x_seq + (size_t)t * BB * HH * WW * CC;
    prep_scores_k<<<BB, 256, 0, stream>>>(x_t, scores, fused, y0a, x0a,
                                          accums + 3, accums + 2);
    prep_stage_k<<<1280, 256, 0, stream>>>(x_t, y0a, x0a, fov16, per16);
    conv_fovea_k<<<BB * 64, 128, 0, stream>>>(fov16, wf16, ff);
    conv_periph_k<<<BB * 256, 64, 0, stream>>>(per16, wp16, pf);
    sal_gate_k<<<BB, 64, 0, stream>>>(ff, gate, accums + 3 + NPATCH);
    lif_fovea_k<<<8192, 256, 0, stream>>>(ff, gate, vf, fused, accums + 0);
    lif_periph_k<<<16384, 256, 0, stream>>>(pf, vp, fused, accums + 1);
    head_wmma_k<<<16, 32, 0, stream>>>(fused, scores, wcat16, head_b, route_b,
                                       out, logits_seq, t);
  }
  finalize_k<<<1, 352, 0, stream>>>(accums, out_tail);
}
